// GraphGRU_64836826301014
// MI455X (gfx1250) — matrix operand order repaired and checked by reading, not compiled, not run on hardware
//
#include <hip/hip_runtime.h>
#include <hip/hip_bf16.h>

typedef __attribute__((ext_vector_type(16))) __bf16 v16bf;
typedef __attribute__((ext_vector_type(8)))  float  v8f;

#define N_MESS 50000
#define MAX_NEI 8
#define INPUT 160
#define HIDDEN 160
#define KTOT 320   // INPUT + HIDDEN

__device__ __forceinline__ float sigmoidf_(float x) { return 1.0f / (1.0f + __expf(-x)); }

__device__ __forceinline__ unsigned packbf2(float a, float b) {
    unsigned short ua = __builtin_bit_cast(unsigned short, (__bf16)a);
    unsigned short ub = __builtin_bit_cast(unsigned short, (__bf16)b);
    return (unsigned)ua | ((unsigned)ub << 16);
}
__device__ __forceinline__ float bfbits2f(unsigned bits16) {
    return __builtin_bit_cast(float, bits16 << 16);
}

// A fragment (16x32 bf16, MxK) for k-step kk from row-major [16][ld] LDS array.
// Lanes 0-15: row M=lane, K = kb..kb+7 (VGPR0-3) and kb+16..kb+23 (VGPR4-7), kb = kk*32.
// Lanes 16-31: same rows, kb offset +8.
__device__ __forceinline__ v16bf loadA_lds(const __bf16* base, int ld, int kk, int lane) {
    int row = lane & 15;
    int kb = kk * 32 + ((lane & 16) ? 8 : 0);
    const __bf16* p = base + row * ld + kb;
    v16bf a;
#pragma unroll
    for (int e = 0; e < 8; ++e) a[e] = p[e];
#pragma unroll
    for (int e = 0; e < 8; ++e) a[8 + e] = p[16 + e];
    return a;
}

// B fragment (32x16 bf16, KxN) from global row-major weight W[OUT][K].
// Column N = lane&15; lanes 0-15 hold K = kb..kb+7 and kb+16..kb+23, lanes 16-31 same with kb offset +8.
__device__ __forceinline__ v16bf loadB_glb(const __bf16* W, int K, int ntile, int kk, int lane) {
    int col = ntile * 16 + (lane & 15);
    int kb = kk * 32 + ((lane & 16) ? 8 : 0);
    const __bf16* p = W + (size_t)col * K + kb;
    v16bf b;
#pragma unroll
    for (int e = 0; e < 8; ++e) b[e] = p[e];
#pragma unroll
    for (int e = 0; e < 8; ++e) b[8 + e] = p[16 + e];
    return b;
}

// vectorized fp32 -> bf16 conversion (n must be a multiple of 4 — true for all our arrays)
__global__ void cvt_bf16_v4(const float4* __restrict__ in, uint2* __restrict__ out, int n4) {
    int i = blockIdx.x * blockDim.x + threadIdx.x;
    if (i < n4) {
        float4 v = in[i];
        uint2 o;
        o.x = packbf2(v.x, v.y);
        o.y = packbf2(v.z, v.w);
        out[i] = o;
    }
}

// out[16-row tile, 160] = inp @ W^T (+ bias). K = 160. One wave per 16-row tile.
// OUTT = float (r1) or __bf16 (Uh, to halve gather traffic).
template <typename OUTT>
__global__ void __launch_bounds__(32) gemm_h160(const float* __restrict__ inp,
                                                const __bf16* __restrict__ W,
                                                const float* __restrict__ bias,
                                                OUTT* __restrict__ out) {
    alignas(16) __shared__ __bf16 a_sh[16 * HIDDEN];
    alignas(16) __shared__ OUTT   out_sh[16 * HIDDEN];
    int row0 = blockIdx.x * 16;
    int lane = threadIdx.x;

    // load 16x160 fp32 tile as float4, pack to bf16 pairs into LDS
    for (int i = lane; i < 16 * HIDDEN / 4; i += 32) {
        float4 v = *(const float4*)(inp + (size_t)row0 * HIDDEN + i * 4);
        uint2 o;
        o.x = packbf2(v.x, v.y);
        o.y = packbf2(v.z, v.w);
        *(uint2*)&a_sh[i * 4] = o;
    }
    __syncthreads();

    int col16 = lane & 15;
    int rbase = (lane & 16) ? 8 : 0;
    for (int nt = 0; nt < 10; ++nt) {
        v8f acc = {};
#pragma unroll
        for (int kk = 0; kk < 5; ++kk) {
            v16bf a = loadA_lds(a_sh, HIDDEN, kk, lane);
            v16bf b = loadB_glb(W, HIDDEN, nt, kk, lane);
            acc = __builtin_amdgcn_wmma_f32_16x16x32_bf16(false, a, false, b,
                                                          (short)0, acc, false, false);
        }
        int col = nt * 16 + col16;
        float bv = bias ? bias[col] : 0.0f;
#pragma unroll
        for (int v = 0; v < 8; ++v)
            out_sh[(rbase + v) * HIDDEN + col] = (OUTT)(acc[v] + bv);
    }
    __syncthreads();

    // contiguous b128 copy-out
    const uint4* src = (const uint4*)out_sh;
    uint4* dst = (uint4*)(out + (size_t)row0 * HIDDEN);
    const int nchunk = (int)(sizeof(OUTT) * 16 * HIDDEN / 16);
    for (int i = lane; i < nchunk; i += 32) dst[i] = src[i];
}

// Fused message pass: vectorized gather, sums, z-gemm (sigmoid), h-gemm (tanh), GRU update.
__global__ void __launch_bounds__(32) gru_msg(const float* __restrict__ h_in,
                                              const __bf16* __restrict__ Uh16,
                                              const float* __restrict__ r1,
                                              const __bf16* __restrict__ x16,
                                              const int* __restrict__ graph,
                                              const __bf16* __restrict__ Wz,
                                              const float* __restrict__ Wz_b,
                                              const __bf16* __restrict__ Wh,
                                              const float* __restrict__ Wh_b,
                                              float* __restrict__ h_out) {
    alignas(16) __shared__ __bf16 act[16 * KTOT];     // [x | sum_h] bf16 A-tile
    alignas(16) __shared__ __bf16 gated[16 * HIDDEN]; // sum_gated bf16
    alignas(16) __shared__ float  sumh[16 * HIDDEN];  // fp32 sum_h, reused as h_out staging
    alignas(16) __shared__ float  zsh[16 * HIDDEN];   // fp32 z
    alignas(16) __shared__ int    nei_sh[16][MAX_NEI];

    int row0 = blockIdx.x * 16;
    int lane = threadIdx.x;

    // stage 128 neighbor indices: exactly 32 int4 chunks, one per lane
    ((int4*)nei_sh)[lane] = ((const int4*)(graph + (size_t)row0 * MAX_NEI))[lane];

    // x part of activation tile: 320 uint4 chunks (8 bf16 each)
    for (int i = lane; i < 16 * (INPUT / 8); i += 32) {
        int r = i / (INPUT / 8), c8 = i - r * (INPUT / 8);
        *(uint4*)&act[r * KTOT + c8 * 8] =
            *(const uint4*)(x16 + (size_t)(row0 + r) * INPUT + c8 * 8);
    }
    __syncthreads();

    // gather + reductions: flat loop over 16 rows x 40 float4-chunks (divergence-free)
    for (int i = lane; i < 16 * (HIDDEN / 4); i += 32) {
        int r = i / (HIDDEN / 4);
        int c = (i - r * (HIDDEN / 4)) * 4;
        int n = row0 + r;
        float4 r1v = *(const float4*)(r1 + (size_t)n * HIDDEN + c);
        float s0 = 0.f, s1 = 0.f, s2 = 0.f, s3 = 0.f;
        float g0 = 0.f, g1 = 0.f, g2 = 0.f, g3 = 0.f;
#pragma unroll
        for (int k = 0; k < MAX_NEI; ++k) {
            int m = nei_sh[r][k];
            float4 hv = *(const float4*)(h_in + (size_t)m * HIDDEN + c);
            uint2 uu = *(const uint2*)(Uh16 + (size_t)m * HIDDEN + c);
            float u0 = bfbits2f(uu.x & 0xffffu), u1 = bfbits2f(uu.x >> 16);
            float u2 = bfbits2f(uu.y & 0xffffu), u3 = bfbits2f(uu.y >> 16);
            s0 += hv.x; s1 += hv.y; s2 += hv.z; s3 += hv.w;
            g0 += sigmoidf_(r1v.x + u0) * hv.x;
            g1 += sigmoidf_(r1v.y + u1) * hv.y;
            g2 += sigmoidf_(r1v.z + u2) * hv.z;
            g3 += sigmoidf_(r1v.w + u3) * hv.w;
        }
        *(float4*)&sumh[r * HIDDEN + c] = make_float4(s0, s1, s2, s3);
        uint2 sp; sp.x = packbf2(s0, s1); sp.y = packbf2(s2, s3);
        *(uint2*)&act[r * KTOT + INPUT + c] = sp;
        uint2 gp; gp.x = packbf2(g0, g1); gp.y = packbf2(g2, g3);
        *(uint2*)&gated[r * HIDDEN + c] = gp;
    }
    __syncthreads();

    int col16 = lane & 15;
    int rbase = (lane & 16) ? 8 : 0;

    // z = sigmoid([x | sum_h] @ Wz^T + b), K = 320
    for (int nt = 0; nt < 10; ++nt) {
        v8f acc = {};
#pragma unroll
        for (int kk = 0; kk < 10; ++kk) {
            v16bf a = loadA_lds(act, KTOT, kk, lane);
            v16bf b = loadB_glb(Wz, KTOT, nt, kk, lane);
            acc = __builtin_amdgcn_wmma_f32_16x16x32_bf16(false, a, false, b,
                                                          (short)0, acc, false, false);
        }
        int col = nt * 16 + col16;
        float bv = Wz_b[col];
#pragma unroll
        for (int v = 0; v < 8; ++v)
            zsh[(rbase + v) * HIDDEN + col] = sigmoidf_(acc[v] + bv);
    }
    __syncthreads();

    // pre_h = tanh([x | sum_gated] @ Wh^T + b); h = ((1-z)*sum_h + z*pre_h) * mask
    for (int nt = 0; nt < 10; ++nt) {
        v8f acc = {};
#pragma unroll
        for (int kk = 0; kk < 10; ++kk) {
            v16bf a = (kk < 5) ? loadA_lds(act, KTOT, kk, lane)          // x region (K 0..159)
                               : loadA_lds(gated, HIDDEN, kk - 5, lane); // gated region
            v16bf b = loadB_glb(Wh, KTOT, nt, kk, lane);
            acc = __builtin_amdgcn_wmma_f32_16x16x32_bf16(false, a, false, b,
                                                          (short)0, acc, false, false);
        }
        int col = nt * 16 + col16;
        float bv = Wh_b[col];
#pragma unroll
        for (int v = 0; v < 8; ++v) {
            int li = (rbase + v) * HIDDEN + col;
            float pre = tanhf(acc[v] + bv);
            float z = zsh[li];
            float s = sumh[li];
            float hv = (1.0f - z) * s + z * pre;
            if (row0 + rbase + v == 0) hv = 0.0f;  // padding-row mask
            sumh[li] = hv;  // restage for coalesced copy-out (read-before-write per element)
        }
    }
    __syncthreads();

    // contiguous b128 copy-out of the 16x160 tile
    const uint4* src = (const uint4*)sumh;
    uint4* dst = (uint4*)(h_out + (size_t)row0 * HIDDEN);
    for (int i = lane; i < 16 * HIDDEN / 4; i += 32) dst[i] = src[i];
}

extern "C" void kernel_launch(void* const* d_in, const int* in_sizes, int n_in,
                              void* d_out, int out_size, void* d_ws, size_t ws_size,
                              hipStream_t stream) {
    (void)in_sizes; (void)n_in; (void)out_size; (void)ws_size;
    const float* h    = (const float*)d_in[0];
    const float* x    = (const float*)d_in[1];
    const int*   gr   = (const int*)d_in[2];
    const float* Wz_w = (const float*)d_in[3];
    const float* Wz_b = (const float*)d_in[4];
    const float* Wr_w = (const float*)d_in[5];
    const float* Ur_w = (const float*)d_in[6];
    const float* Ur_b = (const float*)d_in[7];
    const float* Wh_w = (const float*)d_in[8];
    const float* Wh_b = (const float*)d_in[9];
    float* out = (float*)d_out;

    char* ws = (char*)d_ws;
    size_t off = 0;
    auto wsalloc = [&](size_t bytes) -> char* {
        char* p = ws + off;
        off = (off + bytes + 255) & ~(size_t)255;
        return p;
    };
    __bf16* x16  = (__bf16*)wsalloc((size_t)N_MESS * INPUT * 2);
    __bf16* Wz16 = (__bf16*)wsalloc((size_t)HIDDEN * KTOT * 2);
    __bf16* Wr16 = (__bf16*)wsalloc((size_t)HIDDEN * INPUT * 2);
    __bf16* Ur16 = (__bf16*)wsalloc((size_t)HIDDEN * HIDDEN * 2);
    __bf16* Wh16 = (__bf16*)wsalloc((size_t)HIDDEN * KTOT * 2);
    float*  r1   = (float*)wsalloc((size_t)N_MESS * HIDDEN * 4);
    __bf16* Uh   = (__bf16*)wsalloc((size_t)N_MESS * HIDDEN * 2);
    float*  hA   = (float*)wsalloc((size_t)N_MESS * HIDDEN * 4);

    // one-time bf16 conversions (vectorized)
    auto cvt = [&](const float* src, __bf16* dst, int n) {
        int n4 = n / 4;
        cvt_bf16_v4<<<(n4 + 255) / 256, 256, 0, stream>>>((const float4*)src, (uint2*)dst, n4);
    };
    cvt(x, x16, N_MESS * INPUT);
    cvt(Wz_w, Wz16, HIDDEN * KTOT);
    cvt(Wr_w, Wr16, HIDDEN * INPUT);
    cvt(Ur_w, Ur16, HIDDEN * HIDDEN);
    cvt(Wh_w, Wh16, HIDDEN * KTOT);

    const int ntiles = N_MESS / 16; // 3125 exactly

    // loop-invariant r1 = x @ Wr^T (no bias), fp32 out
    gemm_h160<float><<<ntiles, 32, 0, stream>>>(x, Wr16, nullptr, r1);

    // 3 GRU iterations; ping-pong: in -> out -> hA -> out
    const float* hcur = h;
    float* houts[3] = {out, hA, out};
    for (int d = 0; d < 3; ++d) {
        // Uh = h @ Ur^T + Ur_b (hoisted out of the neighbor loop), bf16 out to halve gather bytes
        gemm_h160<__bf16><<<ntiles, 32, 0, stream>>>(hcur, Ur16, Ur_b, Uh);
        gru_msg<<<ntiles, 32, 0, stream>>>(hcur, Uh, r1, x16, gr,
                                           Wz16, Wz_b, Wh16, Wh_b, houts[d]);
        hcur = houts[d];
    }
}
